// EncDecAD_1408749273271
// MI455X (gfx1250) — matrix operand order repaired and checked by reading, not compiled, not run on hardware
//
#include <hip/hip_runtime.h>

// ---------------------------------------------------------------- constants
constexpr int T_   = 256;
constexpr int B_   = 256;
constexpr int NI_  = 64;
constexpr int NH_  = 512;
constexpr int NO_  = 64;
constexpr int L_   = 2;
constexpr int G4H_ = 4 * NH_;   // 2048

// ---------------------------------------------------------------- types
typedef float  v8f   __attribute__((ext_vector_type(8)));
typedef __bf16 v8bf  __attribute__((ext_vector_type(8)));
typedef __bf16 v16bf __attribute__((ext_vector_type(16)));

union ABf  { v16bf v; v8bf h[2]; };
union BfBits { unsigned short u; __bf16 b; };

static __device__ __forceinline__ __bf16 f2bf(float f) {
  unsigned u = __builtin_bit_cast(unsigned, f);
  u = (u + 0x7FFFu + ((u >> 16) & 1u)) >> 16;   // round-to-nearest-even
  BfBits r; r.u = (unsigned short)u;
  return r.b;
}

static __device__ __forceinline__ float sigmoidf_(float x) {
  return 1.0f / (1.0f + __expf(-x));
}

// ---------------------------------------------------------------- utility kernels
__global__ void cvt_f32_bf16(const float* __restrict__ in, __bf16* __restrict__ out, int n) {
  int i = blockIdx.x * blockDim.x + threadIdx.x;
  if (i < n) out[i] = f2bf(in[i]);
}

__global__ void fill_zero(float* __restrict__ p, int n) {
  int i = blockIdx.x * blockDim.x + threadIdx.x;
  if (i < n) p[i] = 0.0f;
}

// ---------------------------------------------------------------- WMMA GEMM (K = 512 fixed)
// C[M,N](f32) = A1[M,512] @ W1[N,512]^T  (+ A2 @ W2^T if dual)
// A*, W* bf16 row-major. Wave tile 32(M) x 64(N); two A fragments share each
// B fragment. K is compile-time so the fully-unrolled k-loop uses only three
// loop-invariant base addresses with immediate offsets -> no inner-loop VALU,
// no WMMA->VALU hazard NOPs, free clause/prefetch scheduling.
// blockDim = 32*wavesPerBlock.  grid = (ceil(N/(waves*64)), M/32)
__global__ void gemm_wmma(const __bf16* __restrict__ A1, const __bf16* __restrict__ W1,
                          const __bf16* __restrict__ A2, const __bf16* __restrict__ W2,
                          float* __restrict__ C, int N, int dual)
{
  constexpr int K = NH_;   // 512
  const int lane = threadIdx.x & 31;
  const int wave = threadIdx.x >> 5;
  const int m0   = blockIdx.y << 5;                               // 32 rows
  const int n0   = (blockIdx.x * (blockDim.x >> 5) + wave) << 6;  // 64 cols
  if (n0 >= N) return;
  const int half = lane >> 4;   // 0: lanes 0-15, 1: lanes 16-31
  const int l15  = lane & 15;

  v8f acc[2][4] = {};
  const int nPass = dual ? 2 : 1;
#pragma unroll 1
  for (int s = 0; s < nPass; ++s) {
    const __bf16* A = s ? A2 : A1;
    const __bf16* W = s ? W2 : W1;
    const __bf16* arow0 = A + (size_t)(m0 + l15) * K + half * 16;
    const __bf16* arow1 = arow0 + 16 * K;
    const __bf16* wbase = W + (size_t)(n0 + l15) * K + half * 16;
#pragma unroll
    for (int k0 = 0; k0 < K; k0 += 32) {
      ABf a0, a1;
      a0.h[0] = *(const v8bf*)(arow0 + k0);        // K = kb .. kb+7
      a0.h[1] = *(const v8bf*)(arow0 + k0 + 8);    // K = kb+8 .. kb+15
      a1.h[0] = *(const v8bf*)(arow1 + k0);
      a1.h[1] = *(const v8bf*)(arow1 + k0 + 8);
      v16bf b[4];
#pragma unroll
      for (int j = 0; j < 4; ++j)
        b[j] = *(const v16bf*)(wbase + j * 16 * K + k0);
#pragma unroll
      for (int j = 0; j < 4; ++j) {
        acc[0][j] = __builtin_amdgcn_wmma_f32_16x16x32_bf16(
            false, a0.v, false, b[j], (short)0, acc[0][j], false, false);
        acc[1][j] = __builtin_amdgcn_wmma_f32_16x16x32_bf16(
            false, a1.v, false, b[j], (short)0, acc[1][j], false, false);
      }
    }
  }
#pragma unroll
  for (int r = 0; r < 2; ++r) {
#pragma unroll
    for (int j = 0; j < 4; ++j) {
      float* cp = C + (size_t)(m0 + 16 * r + 8 * half) * N + n0 + j * 16 + l15;
#pragma unroll
      for (int v = 0; v < 8; ++v) cp[(size_t)v * N] = acc[r][j][v];
    }
  }
}

// ---------------------------------------------------------------- fused embedding
// emb[T*B,NH] = ((x @ Wi^T + bi) @ We^T + be) as bf16, staging t1 tile in LDS.
// 1 block = 16 rows, 256 threads (8 waves, each owns 64 output columns).
__global__ void emb_fused(const __bf16* __restrict__ xbf, const __bf16* __restrict__ Wi,
                          const float* __restrict__ bi, const __bf16* __restrict__ We,
                          const float* __restrict__ be, __bf16* __restrict__ emb)
{
  __shared__ __bf16 t1[16 * NH_];    // 16 KB
  const int lane = threadIdx.x & 31;
  const int wave = threadIdx.x >> 5;
  const int half = lane >> 4;
  const int l15  = lane & 15;
  const int m0   = blockIdx.x << 4;
  const int n0   = wave << 6;

  { // stage 1: t1 = x @ Wi^T + bi   (K = NI_ = 64)
    v8f acc[4] = {};
    const __bf16* arow  = xbf + (size_t)(m0 + l15) * NI_ + half * 16;
    const __bf16* wbase = Wi + (size_t)(n0 + l15) * NI_ + half * 16;
#pragma unroll
    for (int k0 = 0; k0 < NI_; k0 += 32) {
      ABf a;
      a.h[0] = *(const v8bf*)(arow + k0);
      a.h[1] = *(const v8bf*)(arow + k0 + 8);
      v16bf b[4];
#pragma unroll
      for (int j = 0; j < 4; ++j)
        b[j] = *(const v16bf*)(wbase + j * 16 * NI_ + k0);
#pragma unroll
      for (int j = 0; j < 4; ++j)
        acc[j] = __builtin_amdgcn_wmma_f32_16x16x32_bf16(
            false, a.v, false, b[j], (short)0, acc[j], false, false);
    }
#pragma unroll
    for (int j = 0; j < 4; ++j) {
      int col = n0 + j * 16 + l15;
#pragma unroll
      for (int v = 0; v < 8; ++v)
        t1[(8 * half + v) * NH_ + col] = f2bf(acc[j][v] + bi[col]);
    }
  }
  __syncthreads();
  { // stage 2: emb = t1 @ We^T + be   (K = NH_ = 512), A fragments from LDS
    v8f acc[4] = {};
    const __bf16* lrow  = &t1[(size_t)l15 * NH_ + half * 16];
    const __bf16* wbase = We + (size_t)(n0 + l15) * NH_ + half * 16;
#pragma unroll
    for (int k0 = 0; k0 < NH_; k0 += 32) {
      ABf a;
      a.h[0] = *(const v8bf*)(lrow + k0);
      a.h[1] = *(const v8bf*)(lrow + k0 + 8);
      v16bf b[4];
#pragma unroll
      for (int j = 0; j < 4; ++j)
        b[j] = *(const v16bf*)(wbase + j * 16 * NH_ + k0);
#pragma unroll
      for (int j = 0; j < 4; ++j)
        acc[j] = __builtin_amdgcn_wmma_f32_16x16x32_bf16(
            false, a.v, false, b[j], (short)0, acc[j], false, false);
    }
#pragma unroll
    for (int j = 0; j < 4; ++j) {
      int col = n0 + j * 16 + l15;
#pragma unroll
      for (int v = 0; v < 8; ++v)
        emb[(size_t)(m0 + 8 * half + v) * NH_ + col] = f2bf(acc[j][v] + be[col]);
    }
  }
}

// ---------------------------------------------------------------- encoder gate
// One block (128 thr) per (l,b): gh = sigmoid(h[l,b,:] . G[l,:]); hxg = gh*h (bf16)
__global__ void enc_gate(const float* __restrict__ h, const float* __restrict__ G,
                         float* __restrict__ gh_out, __bf16* __restrict__ hxg)
{
  const int lb = blockIdx.x;        // l*B_ + b
  const int l  = lb >> 8;           // B_ == 256
  const float* hr = h + (size_t)lb * NH_;
  const float* Gl = G + (size_t)l * NH_;
  __shared__ float red[128];
  __shared__ float ghs;
  float s = 0.0f;
  for (int i = threadIdx.x; i < NH_; i += 128) s += hr[i] * Gl[i];
  red[threadIdx.x] = s;
  __syncthreads();
  for (int st = 64; st > 0; st >>= 1) {
    if ((int)threadIdx.x < st) red[threadIdx.x] += red[threadIdx.x + st];
    __syncthreads();
  }
  if (threadIdx.x == 0) {
    float g = sigmoidf_(red[0]);
    ghs = g;
    gh_out[lb] = g;
  }
  __syncthreads();
  const float g = ghs;
  __bf16* hx = hxg + (size_t)lb * NH_;
  for (int i = threadIdx.x; i < NH_; i += 128) hx[i] = f2bf(g * hr[i]);
}

// ---------------------------------------------------------------- decoder gate (divide)
__global__ void dec_div(const float* __restrict__ h, const float* __restrict__ gh,
                        __bf16* __restrict__ hxg)
{
  int idx = blockIdx.x * blockDim.x + threadIdx.x;   // < L_*B_*NH_
  int lb  = idx / NH_;
  hxg[idx] = f2bf(h[idx] / gh[lb]);
}

// ---------------------------------------------------------------- LSTM cell (one layer)
__global__ void lstm_cell(const float* __restrict__ gates, float* __restrict__ h,
                          float* __restrict__ c, __bf16* __restrict__ hbf)
{
  int idx = blockIdx.x * blockDim.x + threadIdx.x;   // < B_*NH_
  int b = idx / NH_, n = idx - b * NH_;
  const float* gr = gates + (size_t)b * G4H_;
  float ig = gr[n];
  float fg = gr[NH_ + n];
  float gg = gr[2 * NH_ + n];
  float og = gr[3 * NH_ + n];
  float cy = sigmoidf_(fg) * c[idx] + sigmoidf_(ig) * tanhf(gg);
  float hy = sigmoidf_(og) * tanhf(cy);
  c[idx] = cy;
  h[idx] = hy;
  hbf[idx] = f2bf(hy);
}

// ---------------------------------------------------------------- host driver
extern "C" void kernel_launch(void* const* d_in, const int* in_sizes, int n_in,
                              void* d_out, int out_size, void* d_ws, size_t ws_size,
                              hipStream_t stream)
{
  (void)in_sizes; (void)n_in; (void)out_size; (void)ws_size;
  const float* x         = (const float*)d_in[0];
  const float* in_lin_W  = (const float*)d_in[1];
  const float* in_lin_b  = (const float*)d_in[2];
  const float* enc_lin_W = (const float*)d_in[3];
  const float* enc_lin_b = (const float*)d_in[4];
  const float* enc_W     = (const float*)d_in[5];
  const float* enc_U     = (const float*)d_in[6];
  const float* enc_G     = (const float*)d_in[7];
  const float* dec_W     = (const float*)d_in[8];
  const float* dec_U     = (const float*)d_in[9];
  const float* dec_L     = (const float*)d_in[10];
  float* out = (float*)d_out;

  char* ws = (char*)d_ws;
  size_t off = 0;
  auto alloc = [&](size_t bytes) -> void* {
    void* p = ws + off;
    off = (off + bytes + 255) & ~(size_t)255;
    return p;
  };

  // ---- workspace layout (~97 MB) ----
  __bf16* x_bf  = (__bf16*)alloc((size_t)T_ * B_ * NI_ * 2);
  __bf16* wInL  = (__bf16*)alloc((size_t)NH_ * NI_ * 2);
  __bf16* wEncL = (__bf16*)alloc((size_t)NH_ * NH_ * 2);
  __bf16* wEncW = (__bf16*)alloc((size_t)L_ * G4H_ * NH_ * 2);
  __bf16* wEncU = (__bf16*)alloc((size_t)L_ * G4H_ * NH_ * 2);
  __bf16* wDecW = (__bf16*)alloc((size_t)(L_ - 1) * G4H_ * NH_ * 2);
  __bf16* wDecU = (__bf16*)alloc((size_t)L_ * G4H_ * NH_ * 2);
  __bf16* wDecL = (__bf16*)alloc((size_t)NO_ * NH_ * 2);
  __bf16* emb   = (__bf16*)alloc((size_t)T_ * B_ * NH_ * 2);
  float*  h     = (float*) alloc((size_t)L_ * B_ * NH_ * 4);
  float*  c     = (float*) alloc((size_t)L_ * B_ * NH_ * 4);
  __bf16* h_bf  = (__bf16*)alloc((size_t)L_ * B_ * NH_ * 2);
  __bf16* hxg   = (__bf16*)alloc((size_t)L_ * B_ * NH_ * 2);
  float*  ghb   = (float*) alloc((size_t)T_ * L_ * B_ * 4);
  float*  gates = (float*) alloc((size_t)B_ * G4H_ * 4);

  // ---- one-time conversions (per call; deterministic) ----
  auto cvt = [&](const float* src, __bf16* dst, int n) {
    cvt_f32_bf16<<<dim3((n + 255) / 256), dim3(256), 0, stream>>>(src, dst, n);
  };
  cvt(x,         x_bf,  T_ * B_ * NI_);
  cvt(in_lin_W,  wInL,  NH_ * NI_);
  cvt(enc_lin_W, wEncL, NH_ * NH_);
  cvt(enc_W,     wEncW, L_ * G4H_ * NH_);
  cvt(enc_U,     wEncU, L_ * G4H_ * NH_);
  cvt(dec_W,     wDecW, (L_ - 1) * G4H_ * NH_);
  cvt(dec_U,     wDecU, L_ * G4H_ * NH_);
  cvt(dec_L,     wDecL, NO_ * NH_);

  fill_zero<<<dim3(L_ * B_ * NH_ / 256), dim3(256), 0, stream>>>(h, L_ * B_ * NH_);
  fill_zero<<<dim3(L_ * B_ * NH_ / 256), dim3(256), 0, stream>>>(c, L_ * B_ * NH_);

  // ---- embedding: emb = (x @ Wi^T + bi) @ We^T + be ----
  emb_fused<<<dim3(T_ * B_ / 16), dim3(256), 0, stream>>>(
      x_bf, wInL, in_lin_b, wEncL, enc_lin_b, emb);

  const dim3 gGrid(4, 8), gBlk(256);             // N=2048 (8 waves*64), M=256/32
  const dim3 oGrid(1, 8), oBlk(32);              // N=64 (1 wave), M=256/32
  const dim3 cellGrid(B_ * NH_ / 256), cellBlk(256);

  // ---- encoder scan ----
  for (int t = 0; t < T_; ++t) {
    enc_gate<<<dim3(L_ * B_), dim3(128), 0, stream>>>(
        h, enc_G, ghb + (size_t)t * L_ * B_, hxg);
    // layer 0: gates = emb_t @ W0^T + hxg0 @ U0^T
    gemm_wmma<<<gGrid, gBlk, 0, stream>>>(
        emb + (size_t)t * B_ * NH_, wEncW,
        hxg, wEncU, gates, G4H_, 1);
    lstm_cell<<<cellGrid, cellBlk, 0, stream>>>(gates, h, c, h_bf);
    // layer 1: gates = h0 @ W1^T + hxg1 @ U1^T
    gemm_wmma<<<gGrid, gBlk, 0, stream>>>(
        h_bf, wEncW + (size_t)G4H_ * NH_,
        hxg + (size_t)B_ * NH_, wEncU + (size_t)G4H_ * NH_, gates, G4H_, 1);
    lstm_cell<<<cellGrid, cellBlk, 0, stream>>>(
        gates, h + (size_t)B_ * NH_, c + (size_t)B_ * NH_, h_bf + (size_t)B_ * NH_);
  }

  // ---- out0 = hT[-1] @ dec_L^T ----
  gemm_wmma<<<oGrid, oBlk, 0, stream>>>(
      h_bf + (size_t)B_ * NH_, wDecL, h_bf + (size_t)B_ * NH_, wDecL,
      out, NO_, 0);

  // ---- decoder scan (T-1 steps, reversed gates) ----
  for (int t = 0; t < T_ - 1; ++t) {
    dec_div<<<dim3(L_ * B_ * NH_ / 256), dim3(256), 0, stream>>>(
        h, ghb + (size_t)(T_ - 2 - t) * L_ * B_, hxg);
    // layer 0: gates = hxg0 @ dec_U0^T (no input term)
    gemm_wmma<<<gGrid, gBlk, 0, stream>>>(
        hxg, wDecU, hxg, wDecU, gates, G4H_, 0);
    lstm_cell<<<cellGrid, cellBlk, 0, stream>>>(gates, h, c, h_bf);
    // layer 1: gates = h0 @ dec_W0^T + hxg1 @ dec_U1^T
    gemm_wmma<<<gGrid, gBlk, 0, stream>>>(
        h_bf, wDecW,
        hxg + (size_t)B_ * NH_, wDecU + (size_t)G4H_ * NH_, gates, G4H_, 1);
    lstm_cell<<<cellGrid, cellBlk, 0, stream>>>(
        gates, h + (size_t)B_ * NH_, c + (size_t)B_ * NH_, h_bf + (size_t)B_ * NH_);
    // out[t+1] = h1 @ dec_L^T
    gemm_wmma<<<oGrid, oBlk, 0, stream>>>(
        h_bf + (size_t)B_ * NH_, wDecL, h_bf + (size_t)B_ * NH_, wDecL,
        out + (size_t)(t + 1) * B_ * NO_, NO_, 0);
  }
}
